// MultiHeadSelfAttention_70866960384366
// MI455X (gfx1250) — matrix operand order repaired and checked by reading, not compiled, not run on hardware
//
#include <hip/hip_runtime.h>

// ---------------- types ----------------
typedef __bf16 bf16;
typedef __attribute__((ext_vector_type(16))) bf16  v16bf;
typedef __attribute__((ext_vector_type(8)))  bf16  v8bf;
typedef __attribute__((ext_vector_type(8)))  float v8f;
typedef __attribute__((ext_vector_type(4)))  unsigned int v4u;
typedef __attribute__((ext_vector_type(8)))  int v8i;
typedef __attribute__((ext_vector_type(4)))  int v4i;

// ---------------- problem constants ----------------
#define BB   2
#define LL   2048
#define DD   1024
#define NH   16
#define HS   64
#define ND3  3072   // 3*D
#define NBLK (LL / 32)

__device__ __forceinline__ v16bf bfcat(v8bf lo, v8bf hi) {
    return __builtin_shufflevector(lo, hi, 0,1,2,3,4,5,6,7,8,9,10,11,12,13,14,15);
}

// A-fragment (16x32 bf16, MxK): lane m = lane&15, koff = (lane>>4)*16.
__device__ __forceinline__ v16bf ldA(const bf16* row, int koff) {
    return bfcat(*(const v8bf*)(row + koff), *(const v8bf*)(row + koff + 8));
}
// B-fragment (32x16 bf16, KxN): lane n = lane&15, ks = (lane>>4)*16.
__device__ __forceinline__ v16bf ldB16(const bf16* p) {
    return bfcat(*(const v8bf*)(p), *(const v8bf*)(p + 8));
}

__device__ __forceinline__ v8f wmma_bf16(v16bf a, v16bf b, v8f c) {
    return __builtin_amdgcn_wmma_f32_16x16x32_bf16(false, a, false, b, (short)0, c, false, false);
}

// ---------------- Tensor Data Mover: 2D bf16 tile -> LDS ----------------
// D# group0: count=1, lds_addr, 57-bit global addr, type=2.
// D# group1: data_size=1 (2B), tensor_dim0/1, tile_dim0/1, dim0 stride.
__device__ __forceinline__ void tdm_load_2d(unsigned lds_off, unsigned long long gaddr,
                                            unsigned dim0, unsigned dim1,
                                            unsigned tile0, unsigned tile1,
                                            unsigned long long stride0) {
    v4u g0;
    g0.x = 1u;                                        // count=1, no gather
    g0.y = lds_off;                                   // lds_addr
    g0.z = (unsigned)(gaddr & 0xFFFFFFFFu);           // global_addr[31:0]
    g0.w = (unsigned)((gaddr >> 32) & 0x01FFFFFFu)    // global_addr[56:32]
         | (2u << 30);                                // type=2 (image)
    v8i g1;
    g1[0] = (int)(1u << 16);                                            // data_size=2B
    g1[1] = (int)((dim0 & 0xFFFFu) << 16);                              // tensor_dim0 lo
    g1[2] = (int)(((dim0 >> 16) & 0xFFFFu) | ((dim1 & 0xFFFFu) << 16)); // dim0 hi | dim1 lo
    g1[3] = (int)(((dim1 >> 16) & 0xFFFFu) | ((tile0 & 0xFFFFu) << 16));// dim1 hi | tile0
    g1[4] = (int)(tile1 & 0xFFFFu);                                     // tile1, tile2=0
    g1[5] = (int)(stride0 & 0xFFFFFFFFu);                               // dim0 stride lo
    g1[6] = (int)((stride0 >> 32) & 0xFFFFu);                           // stride hi, dim1 stride=0
    g1[7] = 0;
    v4i z4 = {0, 0, 0, 0};
#if __clang_major__ >= 23
    v8i z8 = {0, 0, 0, 0, 0, 0, 0, 0};
    __builtin_amdgcn_tensor_load_to_lds(g0, g1, z4, z4, z8, 0);
#else
    __builtin_amdgcn_tensor_load_to_lds(g0, g1, z4, z4, 0);
#endif
}

// =====================================================================
// Kernel 1: QKV projection  (x @ W + in_bias), fp32 in -> bf16 out.
// Workgroup = 256 threads (8 waves). WG tile: M=16, N=128 (wave: 16x16).
// Writes Q pre-scaled by 1/sqrt(64), K row-major, V transposed (hs, L).
// =====================================================================
__global__ __launch_bounds__(256)
void qkv_kernel(const float* __restrict__ x, const float* __restrict__ w,
                const float* __restrict__ in_bias,
                bf16* __restrict__ Qb, bf16* __restrict__ Kb, bf16* __restrict__ Vt) {
    __shared__ bf16 ldsA[16 * 32];     // x tile, [m][k]
    __shared__ bf16 ldsB[128 * 32];    // W tile, transposed [n][k]

    const int t    = threadIdx.x;
    const int lane = t & 31;
    const int wave = t >> 5;
    const int m0   = blockIdx.y * 16;          // row base in (B*L)
    const int nb   = blockIdx.x * 128;         // col base in 3D

    v8f acc = {};

    for (int kk = 0; kk < DD; kk += 32) {
        if (t < 128) {
            const int row = t >> 3, c4 = t & 7;
            const float4 f = *(const float4*)(x + (size_t)(m0 + row) * DD + kk + c4 * 4);
            const int base = row * 32 + c4 * 4;
            ldsA[base + 0] = (bf16)f.x; ldsA[base + 1] = (bf16)f.y;
            ldsA[base + 2] = (bf16)f.z; ldsA[base + 3] = (bf16)f.w;
        }
        #pragma unroll
        for (int i = 0; i < 4; ++i) {
            const int c   = t + 256 * i;
            const int row = c >> 5, c4 = c & 31;
            const float4 f = *(const float4*)(w + (size_t)(kk + row) * ND3 + nb + c4 * 4);
            const int n = c4 * 4;
            ldsB[(n + 0) * 32 + row] = (bf16)f.x;
            ldsB[(n + 1) * 32 + row] = (bf16)f.y;
            ldsB[(n + 2) * 32 + row] = (bf16)f.z;
            ldsB[(n + 3) * 32 + row] = (bf16)f.w;
        }
        __syncthreads();

        const int koff = (lane >> 4) << 4;
        const v16bf afrag = ldA(&ldsA[(lane & 15) * 32], koff);
        const int ks = (lane >> 4) << 4;
        const v16bf bfrag = ldB16(&ldsB[(wave * 16 + (lane & 15)) * 32 + ks]);
        acc = wmma_bf16(afrag, bfrag, acc);
        __syncthreads();
    }

    const int nabs = nb + wave * 16 + (lane & 15);
    const int head = nabs / 192;
    const int r3   = nabs % 192;
    const int part = r3 >> 6;       // 0=Q 1=K 2=V
    const int dd   = r3 & 63;
    const float bias = in_bias[nabs];
    #pragma unroll
    for (int r = 0; r < 8; ++r) {
        const int mabs = m0 + r + ((lane >> 4) << 3);
        const int b  = mabs >> 11;
        const int l  = mabs & 2047;
        const float v = acc[r] + bias;
        const size_t bh = (size_t)(b * NH + head);
        if (part == 0)      Qb[(bh * LL + l) * HS + dd] = (bf16)(v * 0.125f);
        else if (part == 1) Kb[(bh * LL + l) * HS + dd] = (bf16)v;
        else                Vt[(bh * HS + dd) * LL + l] = (bf16)v;
    }
}

// =====================================================================
// Kernel 2: fused flash attention. All 8 waves of a WG share (b,h) and
// walk the same k-blocks, so K/V blocks are staged ONCE per WG via the
// Tensor Data Mover (double-buffered), synchronized with s_wait_tensorcnt
// + workgroup barriers. ALiBi + gamma*adj bias, post-softmax masking.
// =====================================================================
__global__ __launch_bounds__(256)
void attn_kernel(const float* __restrict__ adj, const unsigned char* __restrict__ mask,
                 const float* __restrict__ out_bias, const float* __restrict__ gamma,
                 const bf16* __restrict__ Qb, const bf16* __restrict__ Kb,
                 const bf16* __restrict__ Vt, float* __restrict__ out) {
    __shared__ bf16 ldsK[2][32 * HS];   // K block: 32 keys x 64 dims (row-major)
    __shared__ bf16 ldsV[2][HS * 32];   // V block: 64 dims x 32 keys (row-major)
    __shared__ bf16 ldsP[8][16 * 32];   // per-wave P tile

    const int t    = threadIdx.x;
    const int lane = t & 31;
    const int wave = t >> 5;
    const int gw   = blockIdx.x * 8 + wave;
    const int qt   = gw & 127;
    const int h    = (gw >> 7) & 15;
    const int b    = gw >> 11;
    const int q0   = qt * 16;
    const size_t bh = (size_t)(b * NH + h);

    const int koff = (lane >> 4) << 4;
    const int ks   = (lane >> 4) << 4;

    const bf16* Qrow = Qb + (bh * LL + q0 + (lane & 15)) * HS;
    const v16bf qf0 = ldA(Qrow, koff);
    const v16bf qf1 = ldA(Qrow + 32, koff);

    const float slope = (h < 8) ? exp2f((float)(-(h + 1))) : 0.0f;
    const float gam   = gamma[h];
    const float* adjb = adj + (size_t)b * LL * LL;
    const unsigned char* maskb = mask + (size_t)b * LL;

    float mrun[8], lrun[8];
    #pragma unroll
    for (int r = 0; r < 8; ++r) { mrun[r] = -3.0e38f; lrun[r] = 0.0f; }
    v8f acc0 = {}, acc1 = {}, acc2 = {}, acc3 = {};
    bf16* myP = &ldsP[wave][0];
    const int qrb = q0 + ((lane >> 4) << 3);

    const unsigned long long kbase = (unsigned long long)(size_t)(Kb + bh * LL * HS);
    const unsigned long long vbase = (unsigned long long)(size_t)(Vt + bh * HS * LL);

    // --- prime both buffers via TDM (wave 0 issues for the whole WG) ---
    if (wave == 0) {
        #pragma unroll
        for (int i = 0; i < 2; ++i) {
            tdm_load_2d((unsigned)(size_t)&ldsK[i][0], kbase + (unsigned long long)(i * 32) * HS * 2,
                        HS, LL, HS, 32, HS);            // 32 K-rows of 64, stride 64
            tdm_load_2d((unsigned)(size_t)&ldsV[i][0], vbase + (unsigned long long)(i * 32) * 2,
                        LL, HS, 32, HS, LL);            // 64 V^T-rows of 32, stride 2048
        }
    }

    for (int i = 0; i < NBLK; ++i) {
        const int kb  = i * 32;
        const int cur = i & 1;
        if (wave == 0) {
            if (i + 1 < NBLK) __builtin_amdgcn_s_wait_tensorcnt((short)2);
            else              __builtin_amdgcn_s_wait_tensorcnt((short)0);
        }
        __syncthreads();                    // buffer `cur` ready for all waves

        // prefetch next adj block while we compute
        __builtin_prefetch(adjb + (size_t)qrb * LL + kb + 32, 0, 0);

        // ---- S = Q @ K^T from LDS-staged K block ----
        const bf16* Kr0 = &ldsK[cur][(lane & 15) * HS];
        const bf16* Kr1 = Kr0 + 16 * HS;
        v8f s0 = {}, s1 = {};
        s0 = wmma_bf16(qf0, ldB16(Kr0 + ks), s0);
        s0 = wmma_bf16(qf1, ldB16(Kr0 + 32 + ks), s0);
        s1 = wmma_bf16(qf0, ldB16(Kr1 + ks), s1);
        s1 = wmma_bf16(qf1, ldB16(Kr1 + 32 + ks), s1);

        const int kc0 = kb + (lane & 15);
        const int kc1 = kc0 + 16;
        const float mk0 = maskb[kc0] ? 1.0f : 0.0f;
        const float mk1 = maskb[kc1] ? 1.0f : 0.0f;

        // ---- bias add + online softmax ----
        #pragma unroll
        for (int r = 0; r < 8; ++r) {
            const int q = qrb + r;
            const float* arow = adjb + (size_t)q * LL;
            const float v0 = s0[r] + slope * (-fabsf((float)(kc0 - q))) + gam * arow[kc0];
            const float v1 = s1[r] + slope * (-fabsf((float)(kc1 - q))) + gam * arow[kc1];
            float mrow = fmaxf(v0, v1);
            mrow = fmaxf(mrow, __shfl_xor(mrow, 8, 32));
            mrow = fmaxf(mrow, __shfl_xor(mrow, 4, 32));
            mrow = fmaxf(mrow, __shfl_xor(mrow, 2, 32));
            mrow = fmaxf(mrow, __shfl_xor(mrow, 1, 32));
            const float mnew  = fmaxf(mrun[r], mrow);
            const float alpha = __expf(mrun[r] - mnew);
            const float p0 = __expf(v0 - mnew);
            const float p1 = __expf(v1 - mnew);
            float rs = p0 + p1;                       // row-sum BEFORE masking
            rs += __shfl_xor(rs, 8, 32); rs += __shfl_xor(rs, 4, 32);
            rs += __shfl_xor(rs, 2, 32); rs += __shfl_xor(rs, 1, 32);
            lrun[r] = lrun[r] * alpha + rs;
            mrun[r] = mnew;
            acc0[r] *= alpha; acc1[r] *= alpha; acc2[r] *= alpha; acc3[r] *= alpha;
            const int row = r + ((lane >> 4) << 3);
            myP[row * 32 + (lane & 15)]      = (bf16)(p0 * mk0);
            myP[row * 32 + (lane & 15) + 16] = (bf16)(p1 * mk1);
        }
        __builtin_amdgcn_wave_barrier();

        // ---- O += P @ V from LDS-staged V block ----
        const v16bf pf = ldA(myP + (lane & 15) * 32, koff);
        const bf16* Vc = &ldsV[cur][0] + (lane & 15) * 32 + ks;
        acc0 = wmma_bf16(pf, ldB16(Vc),            acc0);
        acc1 = wmma_bf16(pf, ldB16(Vc + 16 * 32),  acc1);
        acc2 = wmma_bf16(pf, ldB16(Vc + 32 * 32),  acc2);
        acc3 = wmma_bf16(pf, ldB16(Vc + 48 * 32),  acc3);

        __syncthreads();                    // all waves done with buffer `cur`

        // ---- refill `cur` with block kb+64 ----
        if (wave == 0 && kb + 64 < LL) {
            tdm_load_2d((unsigned)(size_t)&ldsK[cur][0],
                        kbase + (unsigned long long)(kb + 64) * HS * 2,
                        HS, LL, HS, 32, HS);
            tdm_load_2d((unsigned)(size_t)&ldsV[cur][0],
                        vbase + (unsigned long long)(kb + 64) * 2,
                        LL, HS, 32, HS, LL);
        }
    }

    // ---- epilogue: mask_i * O / l + out_bias ----
    const float* ob = out_bias + h * 64;
    #pragma unroll
    for (int r = 0; r < 8; ++r) {
        const int q = qrb + r;
        const float mi  = maskb[q] ? 1.0f : 0.0f;
        const float inv = mi / lrun[r];
        float* orow = out + ((size_t)(b * LL + q)) * DD + h * 64 + (lane & 15);
        orow[0]  = acc0[r] * inv + ob[(lane & 15)];
        orow[16] = acc1[r] * inv + ob[16 + (lane & 15)];
        orow[32] = acc2[r] * inv + ob[32 + (lane & 15)];
        orow[48] = acc3[r] * inv + ob[48 + (lane & 15)];
    }
}

// =====================================================================
extern "C" void kernel_launch(void* const* d_in, const int* in_sizes, int n_in,
                              void* d_out, int out_size, void* d_ws, size_t ws_size,
                              hipStream_t stream) {
    (void)in_sizes; (void)n_in; (void)out_size; (void)ws_size;
    const float*         x        = (const float*)d_in[0];
    const float*         adj      = (const float*)d_in[1];
    const unsigned char* mask     = (const unsigned char*)d_in[2];
    const float*         w        = (const float*)d_in[3];
    const float*         in_bias  = (const float*)d_in[4];
    const float*         out_bias = (const float*)d_in[5];
    const float*         gamma    = (const float*)d_in[6];
    float*               out      = (float*)d_out;

    bf16* Qb = (bf16*)d_ws;                          // (B,NH,L,HS) bf16
    bf16* Kb = Qb + (size_t)BB * NH * LL * HS;       // (B,NH,L,HS) bf16
    bf16* Vt = Kb + (size_t)BB * NH * LL * HS;       // (B,NH,HS,L) bf16

    dim3 g1(ND3 / 128, (BB * LL) / 16);              // (24, 256)
    qkv_kernel<<<g1, 256, 0, stream>>>(x, w, in_bias, Qb, Kb, Vt);

    const int nwaves = BB * NH * (LL / 16);          // 4096 q-tile waves
    attn_kernel<<<nwaves / 8, 256, 0, stream>>>(adj, mask, out_bias, gamma,
                                                Qb, Kb, Vt, out);
}
